// RandomWalkEmbed_35751307772374
// MI455X (gfx1250) — matrix operand order repaired and checked by reading, not compiled, not run on hardware
//
#include <hip/hip_runtime.h>
#include <hip/hip_bf16.h>

#define FEAT   2048
#define NP     64
#define NG     256
#define NPAIR  (NP * NG)     // 16384
#define NCLS   751
#define NCLSP  768           // padded to 48 tiles of 16
#define BN_EPS 1e-5f
#define KCHUNK 128           // K staged per LDS buffer
#define NKO    (FEAT / KCHUNK)

typedef __attribute__((ext_vector_type(16))) __bf16 bf16x16;
typedef __attribute__((ext_vector_type(8)))  float  floatx8;

// ---------------------------------------------------------------------------
// Kernel 1: exact batch statistics of d[n,f] = (p_i[f]-g_j[f])^2 via moments.
// ---------------------------------------------------------------------------
__global__ void rw_stats_kernel(const float* __restrict__ probe,
                                const float* __restrict__ gallery,
                                const float* __restrict__ gamma,
                                float* __restrict__ meanv,
                                float* __restrict__ scalev)
{
    int f = blockIdx.x * blockDim.x + threadIdx.x;
    if (f >= FEAT) return;
    float sp1 = 0.f, sp2 = 0.f, sp3 = 0.f, sp4 = 0.f;
    for (int i = 0; i < NP; ++i) {
        float x = probe[i * FEAT + f];
        float x2 = x * x;
        sp1 += x; sp2 += x2; sp3 += x2 * x; sp4 += x2 * x2;
    }
    float sg1 = 0.f, sg2 = 0.f, sg3 = 0.f, sg4 = 0.f;
    for (int j = 0; j < NG; ++j) {
        float x = gallery[j * FEAT + f];
        float x2 = x * x;
        sg1 += x; sg2 += x2; sg3 += x2 * x; sg4 += x2 * x2;
    }
    const float invN = 1.0f / (float)NPAIR;
    float mean = ((float)NG * sp2 + (float)NP * sg2 - 2.f * sp1 * sg1) * invN;
    float m2   = ((float)NG * sp4 - 4.f * sp3 * sg1 + 6.f * sp2 * sg2
                  - 4.f * sp1 * sg3 + (float)NP * sg4) * invN;
    float var  = m2 - mean * mean;
    meanv[f]  = mean;
    scalev[f] = gamma[f] * rsqrtf(var + BN_EPS);
}

// ---------------------------------------------------------------------------
// Kernel 2: fold BN into classifier.
//   W'[c,f]  = bf16( W[c,f] * scale[f] )            (padded to 768 rows)
//   bias2[c] = b[c] + sum_f (beta[f] - mean[f]*scale[f]) * W[c,f]
// ---------------------------------------------------------------------------
__global__ void rw_fold_kernel(const float* __restrict__ W,
                               const float* __restrict__ bias,
                               const float* __restrict__ beta,
                               const float* __restrict__ meanv,
                               const float* __restrict__ scalev,
                               __bf16* __restrict__ wbf,
                               float* __restrict__ bias2)
{
    __shared__ float red[256];
    const int c = blockIdx.x;     // 0..767
    const int t = threadIdx.x;    // 0..255
    float acc = 0.f;
    if (c < NCLS) {
        for (int f = t; f < FEAT; f += 256) {
            float w = W[c * FEAT + f];
            float s = scalev[f];
            wbf[c * FEAT + f] = (__bf16)(w * s);
            acc += (beta[f] - meanv[f] * s) * w;
        }
    } else {
        for (int f = t; f < FEAT; f += 256) wbf[c * FEAT + f] = (__bf16)0.f;
    }
    red[t] = acc;
    __syncthreads();
    for (int off = 128; off > 0; off >>= 1) {
        if (t < off) red[t] += red[t + off];
        __syncthreads();
    }
    if (t == 0) bias2[c] = (c < NCLS) ? bias[c] + red[0] : 0.f;
}

// ---------------------------------------------------------------------------
// Kernel 3: WMMA GEMM with async-to-LDS staged B and register-double-buffered
// B fragments.   out[n,c] = sum_f (p_i[f]-g_j[f])^2 * W'[c,f] + bias2[c]
// Block = 4 waves, block tile 64M x 64N. B tile (64 x 128K bf16 = 16KB)
// staged double-buffered in LDS via GLOBAL_LOAD_ASYNC_TO_LDS_B128 (ASYNCcnt);
// A built on the fly from probe/gallery. 4 WMMAs per 32-K step.
// ---------------------------------------------------------------------------
__global__ __launch_bounds__(128)
void rw_gemm_kernel(const float* __restrict__ probe,
                    const float* __restrict__ gallery,
                    const __bf16* __restrict__ wbf,
                    const float* __restrict__ bias2,
                    float* __restrict__ out)
{
    __shared__ __bf16 Bs[2 * 64 * KCHUNK];   // 32 KB double buffer

    const int tid    = threadIdx.x;
    const int lane   = tid & 31;
    const int wid    = tid >> 5;
    const int mtile  = blockIdx.y * 4 + wid;   // 0..1023
    const int nbase  = mtile * 16;             // first pair-row of tile
    const int iProbe = nbase >> 8;             // 4 waves share one probe row
    const int jbase  = nbase & 255;            // 16 consecutive gallery rows
    const int col16  = lane & 15;
    const int kg     = lane >> 4;              // K-group (0 or 1)

    const float* prow = probe   + (size_t)iProbe * FEAT;
    const float* grow = gallery + (size_t)(jbase + col16) * FEAT;
    const int cbase = blockIdx.x * 64;

    // Async staging plan: 1024 16B chunks; thread covers rows (tid>>4)+8q,
    // 16B-chunk (tid&15) within a 256B row; 8 issues per thread per KCHUNK.
    const unsigned rowbyte0 =
        (unsigned)(((cbase + (tid >> 4)) * FEAT) * 2 + (tid & 15) * 16);
    const unsigned ldsbase = (unsigned)(size_t)(&Bs[0]) + (unsigned)(tid * 16);

    floatx8 acc0 = {}, acc1 = {}, acc2 = {}, acc3 = {};

    // prologue: stage K-chunk 0 into buffer 0
#pragma unroll
    for (int q = 0; q < 8; ++q) {
        unsigned goff = rowbyte0 + (unsigned)(q * (8 * FEAT * 2));
        unsigned loff = ldsbase + (unsigned)(q * 2048);
        asm volatile("global_load_async_to_lds_b128 %0, %1, %2"
                     :: "v"(loff), "v"(goff), "s"(wbf) : "memory");
    }

    for (int ko = 0; ko < NKO; ++ko) {
        const int cur = ko & 1;
        if (ko + 1 < NKO) {
            const unsigned kcb = (unsigned)((ko + 1) * KCHUNK * 2);
            const unsigned lb  = ldsbase + (unsigned)((cur ^ 1) * 16384);
#pragma unroll
            for (int q = 0; q < 8; ++q) {
                unsigned goff = rowbyte0 + (unsigned)(q * (8 * FEAT * 2)) + kcb;
                unsigned loff = lb + (unsigned)(q * 2048);
                asm volatile("global_load_async_to_lds_b128 %0, %1, %2"
                             :: "v"(loff), "v"(goff), "s"(wbf) : "memory");
            }
            // current buffer's 8 asyncs done; prefetch's 8 still in flight
            asm volatile("s_wait_asynccnt 0x8" ::: "memory");
        } else {
            asm volatile("s_wait_asynccnt 0x0" ::: "memory");
        }
        __syncthreads();   // all waves' staging portions visible

        // per-column LDS pointers (row-major 64 x KCHUNK, element -> kg*16+e)
        const __bf16* bsrc  = &Bs[(size_t)cur * (64 * KCHUNK)];
        const __bf16* bcol0 = bsrc + ( 0 + col16) * KCHUNK + kg * 16;
        const __bf16* bcol1 = bsrc + (16 + col16) * KCHUNK + kg * 16;
        const __bf16* bcol2 = bsrc + (32 + col16) * KCHUNK + kg * 16;
        const __bf16* bcol3 = bsrc + (48 + col16) * KCHUNK + kg * 16;

        // register double-buffer: preload K-step 0's B fragments
        bf16x16 nb0 = *(const bf16x16*)(bcol0);
        bf16x16 nb1 = *(const bf16x16*)(bcol1);
        bf16x16 nb2 = *(const bf16x16*)(bcol2);
        bf16x16 nb3 = *(const bf16x16*)(bcol3);

#pragma unroll
        for (int ki = 0; ki < KCHUNK / 32; ++ki) {
            const int k0 = ko * KCHUNK + ki * 32;

            bf16x16 b0 = nb0, b1 = nb1, b2 = nb2, b3 = nb3;
            if (ki + 1 < KCHUNK / 32) {
                // issue next step's DS loads early; A build hides their latency
                nb0 = *(const bf16x16*)(bcol0 + (ki + 1) * 32);
                nb1 = *(const bf16x16*)(bcol1 + (ki + 1) * 32);
                nb2 = *(const bf16x16*)(bcol2 + (ki + 1) * 32);
                nb3 = *(const bf16x16*)(bcol3 + (ki + 1) * 32);
            }

            // ---- A fragment: elements 0..15 -> K = kg*16 + 0..15
            const float* pa = prow + k0 + kg * 16;
            const float* ga = grow + k0 + kg * 16;
            float4 p0 = *(const float4*)(pa +  0);
            float4 p1 = *(const float4*)(pa +  4);
            float4 p2 = *(const float4*)(pa +  8);
            float4 p3 = *(const float4*)(pa + 12);
            float4 g0 = *(const float4*)(ga +  0);
            float4 g1 = *(const float4*)(ga +  4);
            float4 g2 = *(const float4*)(ga +  8);
            float4 g3 = *(const float4*)(ga + 12);

            bf16x16 a;
            float d;
            d = p0.x - g0.x; a[ 0] = (__bf16)(d * d);
            d = p0.y - g0.y; a[ 1] = (__bf16)(d * d);
            d = p0.z - g0.z; a[ 2] = (__bf16)(d * d);
            d = p0.w - g0.w; a[ 3] = (__bf16)(d * d);
            d = p1.x - g1.x; a[ 4] = (__bf16)(d * d);
            d = p1.y - g1.y; a[ 5] = (__bf16)(d * d);
            d = p1.z - g1.z; a[ 6] = (__bf16)(d * d);
            d = p1.w - g1.w; a[ 7] = (__bf16)(d * d);
            d = p2.x - g2.x; a[ 8] = (__bf16)(d * d);
            d = p2.y - g2.y; a[ 9] = (__bf16)(d * d);
            d = p2.z - g2.z; a[10] = (__bf16)(d * d);
            d = p2.w - g2.w; a[11] = (__bf16)(d * d);
            d = p3.x - g3.x; a[12] = (__bf16)(d * d);
            d = p3.y - g3.y; a[13] = (__bf16)(d * d);
            d = p3.z - g3.z; a[14] = (__bf16)(d * d);
            d = p3.w - g3.w; a[15] = (__bf16)(d * d);

            acc0 = __builtin_amdgcn_wmma_f32_16x16x32_bf16(false, a, false, b0,
                                                           (short)0, acc0, false, false);
            acc1 = __builtin_amdgcn_wmma_f32_16x16x32_bf16(false, a, false, b1,
                                                           (short)0, acc1, false, false);
            acc2 = __builtin_amdgcn_wmma_f32_16x16x32_bf16(false, a, false, b2,
                                                           (short)0, acc2, false, false);
            acc3 = __builtin_amdgcn_wmma_f32_16x16x32_bf16(false, a, false, b3,
                                                           (short)0, acc3, false, false);
        }
        __syncthreads();   // done reading buf[cur] before it is re-staged
    }

    // ---- Epilogue: D element r -> row M = r + 8*kg, column N = col16
    const int rowbase = nbase + 8 * kg;
#pragma unroll
    for (int t = 0; t < 4; ++t) {
        int c = cbase + t * 16 + col16;
        if (c >= NCLS) continue;
        float bb = bias2[c];
        floatx8 accv = (t == 0) ? acc0 : (t == 1) ? acc1 : (t == 2) ? acc2 : acc3;
#pragma unroll
        for (int r = 0; r < 8; ++r) {
            out[(size_t)(rowbase + r) * NCLS + c] = accv[r] + bb;
        }
    }
}

extern "C" void kernel_launch(void* const* d_in, const int* in_sizes, int n_in,
                              void* d_out, int out_size, void* d_ws, size_t ws_size,
                              hipStream_t stream) {
    const float* probe   = (const float*)d_in[0];  // [64, 2048]
    const float* gallery = (const float*)d_in[1];  // [256, 2048]
    const float* gamma   = (const float*)d_in[2];  // [2048]
    const float* beta    = (const float*)d_in[3];  // [2048]
    const float* W       = (const float*)d_in[4];  // [751, 2048]
    const float* b       = (const float*)d_in[5];  // [751]
    float* out = (float*)d_out;                    // [64, 256, 751]

    // workspace (~3.2 MB): mean | scale | bias2(pad 768) | W' bf16 [768][2048]
    float*  meanv  = (float*)d_ws;
    float*  scalev = meanv + FEAT;
    float*  bias2  = scalev + FEAT;
    __bf16* wbf    = (__bf16*)(bias2 + NCLSP);

    rw_stats_kernel<<<dim3(FEAT / 256), dim3(256), 0, stream>>>(
        probe, gallery, gamma, meanv, scalev);

    rw_fold_kernel<<<dim3(NCLSP), dim3(256), 0, stream>>>(
        W, b, beta, meanv, scalev, wbf, bias2);

    // grid: x = 12 tiles of 64 classes, y = 256 blocks * 4 waves = 1024 M-tiles
    rw_gemm_kernel<<<dim3(NCLSP / 64, NPAIR / 16 / 4), dim3(128), 0, stream>>>(
        probe, gallery, wbf, bias2, out);
}
